// SpiderPC_21715354650021
// MI455X (gfx1250) — matrix operand order repaired and checked by reading, not compiled, not run on hardware
//
#include <hip/hip_runtime.h>

// ---------------------------------------------------------------------------
// SpiderCNN-style point network on gfx1250 (MI455X), wave32 + WMMA.
// - kNN/top-k/selection: wave32 VALU + LDS wave-argmax merge (selection-bound)
// - bn2d folded analytically via 11 global moments (inputs are affine in them)
// - FLOP-dominant projections + MLP tail: v_wmma_f32_16x16x32_f16, f32 accum,
//   A/B fragments per ISA 7.12.2 lane layout; branchless hot loop so the
//   compiler software-pipelines loads across wmma (partial s_wait_loadcnt);
//   global_prefetch_b8 warm-up hoisted out of the loop.
// ---------------------------------------------------------------------------

typedef __attribute__((ext_vector_type(16))) _Float16 v16h;
typedef __attribute__((ext_vector_type(8)))  _Float16 v8h;
typedef __attribute__((ext_vector_type(8)))  float    v8f;

#define BB   4
#define NN   4096
#define KNN  20
#define EPSV 1e-5f

// ============================== sa = |x|^2 ==================================
__global__ void sa_kernel(const float* __restrict__ X, float* __restrict__ SA) {
  int gid = blockIdx.x * 256 + threadIdx.x;          // B*N
  int b = gid >> 12, n = gid & (NN - 1);
  const float* xb = X + (size_t)b * 3 * NN;
  float x0 = xb[n], x1 = xb[NN + n], x2 = xb[2 * NN + n];
  SA[gid] = x0 * x0 + x1 * x1 + x2 * x2;
}

// ===================== kNN (top-20 of -sqdist) + inv density ================
// one wave per query row; per-lane sorted top-20; wave-argmax merge via LDS.
__global__ void knn_kernel(const float* __restrict__ X, const float* __restrict__ SA,
                           int* __restrict__ NNI, float* __restrict__ INVD) {
  __shared__ float lv  [8][32][KNN];
  __shared__ int   lixs[8][32][KNN];
  int wave = threadIdx.x >> 5, lane = threadIdx.x & 31;
  int row  = blockIdx.x * 8 + wave;                  // 0 .. B*N-1
  int b = row >> 12, m = row & (NN - 1);
  const float* xb  = X + (size_t)b * 3 * NN;
  const float* sab = SA + b * NN;
  float xm0 = xb[m], xm1 = xb[NN + m], xm2 = xb[2 * NN + m];
  float sam = sab[m];

  float vals[KNN]; int ids[KNN];
#pragma unroll
  for (int k = 0; k < KNN; ++k) { vals[k] = -3.0e38f; ids[k] = 0; }

  for (int n = lane; n < NN; n += 32) {
    float dot = xm0 * xb[n] + xm1 * xb[NN + n] + xm2 * xb[2 * NN + n];
    float v = 2.f * dot - sam - sab[n];              // negd
    if (v > vals[KNN - 1]) {
      vals[KNN - 1] = v; ids[KNN - 1] = n;
#pragma unroll
      for (int j = KNN - 1; j > 0; --j) {
        if (vals[j] > vals[j - 1]) {
          float tv = vals[j]; vals[j] = vals[j - 1]; vals[j - 1] = tv;
          int   ti = ids[j];  ids[j]  = ids[j - 1];  ids[j - 1]  = ti;
        }
      }
    }
  }
#pragma unroll
  for (int k = 0; k < KNN; ++k) { lv[wave][lane][k] = vals[k]; lixs[wave][lane][k] = ids[k]; }
  __syncthreads();

  int cur = 0;
  float gsum = 0.f;
  int* outp = NNI + (size_t)row * KNN;
  for (int r = 0; r < KNN; ++r) {
    int cc = (cur < KNN) ? cur : (KNN - 1);
    float cv = lv[wave][lane][cc];
    int   ci = lixs[wave][lane][cc];
    if (cur >= KNN) cv = -3.0e38f;
    float bv = cv; int bl = lane; int bi = ci;
#pragma unroll
    for (int o = 16; o >= 1; o >>= 1) {
      float ov = __shfl_xor(bv, o, 32);
      int   ol = __shfl_xor(bl, o, 32);
      int   oi = __shfl_xor(bi, o, 32);
      if (ov > bv || (ov == bv && oi < bi)) { bv = ov; bl = ol; bi = oi; }
    }
    if (lane == bl) cur++;
    gsum += __expf(2.0f * bv);                       // exp(vals/(2*sigma^2)), sigma=0.5
    if (lane == 0) outp[r] = bi;
  }
  if (lane == 0) INVD[row] = (float)KNN / gsum;
}

// =========== scale_raw, lnn_pos, + per-block partial global moments =========
__global__ void prep_kernel(const float* __restrict__ X, const int* __restrict__ NNI,
                            const float* __restrict__ INVD, float* __restrict__ SRW,
                            float* __restrict__ LNN, float* __restrict__ MOMP) {
  int tid = threadIdx.x;
  int gid = blockIdx.x * 256 + tid;                  // B*N
  int b = gid >> 12, n = gid & (NN - 1);
  const int* idp = NNI + (size_t)gid * KNN;
  int id[KNN]; float iv[KNN]; float mx = 0.f;
#pragma unroll
  for (int k = 0; k < KNN; ++k) {
    id[k] = idp[k];
    iv[k] = INVD[b * NN + id[k]];
    mx = fmaxf(mx, iv[k]);
  }
  float p[11];
#pragma unroll
  for (int j = 0; j < 11; ++j) p[j] = 0.f;
  float inv = 1.f / mx;
#pragma unroll
  for (int k = 0; k < KNN; ++k) {
    float s = iv[k] * inv;
    SRW[((size_t)(b * KNN + k)) * NN + n] = s;
    p[0] += s; p[1] += s * s;
  }
  const float* xb = X + (size_t)b * 3 * NN;
  float c0 = xb[n], c1 = xb[NN + n], c2 = xb[2 * NN + n];
#pragma unroll
  for (int k = 0; k < KNN; ++k) {
    float l0 = xb[id[k]]          - c0;
    float l1 = xb[NN + id[k]]     - c1;
    float l2 = xb[2 * NN + id[k]] - c2;
    LNN[((size_t)((b * 3 + 0) * KNN + k)) * NN + n] = l0;
    LNN[((size_t)((b * 3 + 1) * KNN + k)) * NN + n] = l1;
    LNN[((size_t)((b * 3 + 2) * KNN + k)) * NN + n] = l2;
    p[2] += l0; p[3] += l1; p[4] += l2;
    p[5] += l0 * l0; p[6] += l0 * l1; p[7] += l0 * l2;
    p[8] += l1 * l1; p[9] += l1 * l2; p[10] += l2 * l2;
  }
  __shared__ float red[256];
  for (int j = 0; j < 11; ++j) {
    red[tid] = p[j]; __syncthreads();
    for (int s = 128; s > 0; s >>= 1) { if (tid < s) red[tid] += red[tid + s]; __syncthreads(); }
    if (tid == 0) MOMP[blockIdx.x * 16 + j] = red[0];
    __syncthreads();
  }
}

__global__ void momsum_kernel(const float* __restrict__ MOMP, float* __restrict__ MOM) {
  int t = threadIdx.x;
  if (t < 11) {
    float s = 0.f;
    for (int blk = 0; blk < 64; ++blk) s += MOMP[blk * 16 + t];
    MOM[t] = s;
  }
}

// ============ fold bn2d analytically into per-channel affine consts =========
struct K3P {
  const float* dn_w1[3]; const float* dn_g[3]; const float* dn_beta[3];
  const float* wn_w[3];  const float* wn_g[3]; const float* wn_beta[3];
};

__global__ void lconst_kernel(const float* __restrict__ MOM, float* __restrict__ LC, K3P P) {
  int t = threadIdx.x;                               // 32 threads
  const float cnt = (float)BB * KNN * NN;
  float mean_s = MOM[0] / cnt;
  float var_s  = MOM[1] / cnt - mean_s * mean_s;
  float md[3];
  md[0] = MOM[2] / cnt; md[1] = MOM[3] / cnt; md[2] = MOM[4] / cnt;
  float cov[3][3];
  cov[0][0] = MOM[5] / cnt - md[0] * md[0];
  cov[0][1] = cov[1][0] = MOM[6] / cnt - md[0] * md[1];
  cov[0][2] = cov[2][0] = MOM[7] / cnt - md[0] * md[2];
  cov[1][1] = MOM[8] / cnt - md[1] * md[1];
  cov[1][2] = cov[2][1] = MOM[9] / cnt - md[1] * md[2];
  cov[2][2] = MOM[10] / cnt - md[2] * md[2];

  for (int li = 0; li < 3; ++li) {
    float* L = LC + li * 256;
    if (t < 16) {
      float w1 = P.dn_w1[li][t];
      float rs = rsqrtf(w1 * w1 * var_s + EPSV);
      float a  = w1 * rs * P.dn_g[li][t];
      L[128 + t] = a;
      L[144 + t] = P.dn_beta[li][t] - a * mean_s;    // dn_b1 cancels through bn mean
    }
    float w[3];
    for (int d = 0; d < 3; ++d) w[d] = P.wn_w[li][t * 3 + d];
    float var = 0.f;
    for (int d = 0; d < 3; ++d)
      for (int e = 0; e < 3; ++e) var += w[d] * w[e] * cov[d][e];
    float rs = rsqrtf(var + EPSV);
    float g  = P.wn_g[li][t];
    float dotm = 0.f;
    for (int d = 0; d < 3; ++d) {
      float A = w[d] * rs * g;
      L[t * 3 + d] = A;
      dotm += A * md[d];
    }
    L[96 + t] = P.wn_beta[li][t] - dotm;             // wn_b cancels through bn mean
  }
}

// ===== per-point feature build: feat[n, c*32+w] = sum_k scaled*wmat (f16) ===
__global__ void feat_kernel(const float* __restrict__ xin, int bstride, int ci,
                            const int* __restrict__ NNI, const float* __restrict__ SRW,
                            const float* __restrict__ LNN, const float* __restrict__ LC,
                            const float* __restrict__ dn_w2, const float* __restrict__ dn_b2,
                            _Float16* __restrict__ Feat) {
  __shared__ float sA[16], sD[16], sW2[16], sWA[96], sWB[32];
  int tid = threadIdx.x;
  if (tid < 16) { sA[tid] = LC[128 + tid]; sD[tid] = LC[144 + tid]; sW2[tid] = dn_w2[tid]; }
  if (tid < 96) sWA[tid] = LC[tid];
  if (tid >= 96 && tid < 128) sWB[tid - 96] = LC[tid];
  __syncthreads();

  int wave = tid >> 5, lane = tid & 31;
  int pt = blockIdx.x * 8 + wave;                    // B*N points
  int b = pt >> 12, n = pt & (NN - 1);
  float b2 = dn_b2[0];
  float A0 = sWA[lane * 3 + 0], A1 = sWA[lane * 3 + 1], A2 = sWA[lane * 3 + 2], Bc = sWB[lane];

  const int* idp = NNI + (size_t)pt * KNN;
  int id[KNN];
#pragma unroll
  for (int k = 0; k < KNN; ++k) id[k] = idp[k];

  float mk[KNN];
#pragma unroll
  for (int k = 0; k < KNN; ++k) {
    float s = SRW[((size_t)(b * KNN + k)) * NN + n];
    float t = b2;
#pragma unroll
    for (int c = 0; c < 16; ++c) t += sW2[c] * fmaxf(sA[c] * s + sD[c], 0.f);
    float dsc = 1.f / (1.f + __expf(-t));            // sigmoid
    float l0 = LNN[((size_t)((b * 3 + 0) * KNN + k)) * NN + n];
    float l1 = LNN[((size_t)((b * 3 + 1) * KNN + k)) * NN + n];
    float l2 = LNN[((size_t)((b * 3 + 2) * KNN + k)) * NN + n];
    float w = fmaxf(A0 * l0 + A1 * l1 + A2 * l2 + Bc, 0.f);
    mk[k] = dsc * w;
  }
  const float* xb = xin + (size_t)b * bstride;
  _Float16* fp = Feat + (size_t)pt * (ci * 32) + lane;
  for (int c = 0; c < ci; ++c) {
    const float* xc = xb + (size_t)c * NN;
    float acc = 0.f;
#pragma unroll
    for (int k = 0; k < KNN; ++k) acc += xc[id[k]] * mk[k];
    fp[c * 32] = (_Float16)acc;
  }
}

// ============================ f32 -> f16 converters =========================
__global__ void w16_kernel(const float* __restrict__ src, _Float16* __restrict__ dst,
                           int rows, int rowsPad, int cols) {
  int gid = blockIdx.x * 256 + threadIdx.x;
  if (gid >= rowsPad * cols) return;
  int r = gid / cols;
  dst[gid] = (r < rows) ? (_Float16)src[gid] : (_Float16)0.f;
}

__global__ void act16_kernel(const float* __restrict__ src, _Float16* __restrict__ dst,
                             int mreal, int kd, int lds) {
  int gid = blockIdx.x * 256 + threadIdx.x;          // 16*kd
  if (gid >= 16 * kd) return;
  int m = gid / kd, k = gid - m * kd;
  dst[gid] = (m < mreal) ? (_Float16)src[(size_t)m * lds + k] : (_Float16)0.f;
}

// ================= WMMA f16 GEMM: out = A(MxK) * W(OxK)^T + bias ============
union HF16 { v16h v; v8h h[2]; };

__global__ void gemm16_kernel(const _Float16* __restrict__ A, const _Float16* __restrict__ W,
                              const float* __restrict__ bias, float* __restrict__ out,
                              int Kd, int Nreal, int Npad, int M, int m_real,
                              int mode, int ldo, int Nloc, int co) {
  int wave = threadIdx.x >> 5, lane = threadIdx.x & 31;
  int tiles_n = Npad >> 4;
  int total = (M >> 4) * tiles_n;
  int tile = blockIdx.x * 8 + wave;
  if (tile >= total) return;                         // uniform per wave
  int tm = tile / tiles_n, tn = tile - tm * tiles_n;

  int row    = tm * 16 + (lane & 15);
  int khalf  = (lane >> 4) * 8;                      // A lane K layout (ISA 7.12.2)
  int ocol   = tn * 16 + (lane & 15);
  int kstart = (lane >> 4) * 8;                      // B lane K layout
  const _Float16* Arow = A + (size_t)row  * Kd;
  const _Float16* Wrow = W + (size_t)ocol * Kd;

  // one-time stream warm-up (global_prefetch_b8); kept OUT of the hot loop so
  // the k-loop stays branchless and the compiler pipelines loads across wmma.
  __builtin_prefetch((const void*)Arow, 0, 1);
  __builtin_prefetch((const void*)Wrow, 0, 1);
  if (Kd > 512) {
    __builtin_prefetch((const void*)(Arow + 512), 0, 1);
    __builtin_prefetch((const void*)(Wrow + 512), 0, 1);
  }

  v8f acc = {};
  for (int kk = 0; kk < Kd; kk += 32) {
    HF16 a, b;
    a.h[0] = *(const v8h*)(Arow + kk + khalf);
    a.h[1] = *(const v8h*)(Arow + kk + 16 + khalf);
    b.h[0] = *(const v8h*)(Wrow + kk + kstart);
    b.h[1] = *(const v8h*)(Wrow + kk + 16 + kstart);
    acc = __builtin_amdgcn_wmma_f32_16x16x32_f16(false, a.v, false, b.v,
                                                 (short)0, acc, false, false);
  }
  float bv = (ocol < Nreal) ? bias[ocol] : 0.f;
#pragma unroll
  for (int r = 0; r < 8; ++r) {
    int mr = tm * 16 + r + ((lane >> 4) * 8);        // D layout: M = r + 8*(lane>=16)
    if (ocol < Nreal && mr < m_real) {
      float val = acc[r] + bv;
      if (mode == 0) {
        out[(size_t)mr * ldo + ocol] = val;
      } else {
        int bb = mr / Nloc, nn = mr - bb * Nloc;
        out[((size_t)(bb * co + ocol)) * Nloc + nn] = val;
      }
    }
  }
}

// ============================== bn1d reduce/apply ===========================
__global__ void bnstat_kernel(const float* __restrict__ outpre, int co, float* __restrict__ BNST) {
  int o = blockIdx.x, tid = threadIdx.x;
  float s = 0.f, s2 = 0.f;
  for (int i = tid; i < BB * NN; i += 256) {
    int b = i >> 12, n = i & (NN - 1);
    float v = outpre[((size_t)(b * co + o)) * NN + n];
    s += v; s2 += v * v;
  }
  __shared__ float r1[256], r2[256];
  r1[tid] = s; r2[tid] = s2; __syncthreads();
  for (int st = 128; st > 0; st >>= 1) {
    if (tid < st) { r1[tid] += r1[tid + st]; r2[tid] += r2[tid + st]; }
    __syncthreads();
  }
  if (tid == 0) { BNST[o * 2] = r1[0]; BNST[o * 2 + 1] = r2[0]; }
}

__global__ void bnapply_kernel(const float* __restrict__ outpre, const float* __restrict__ BNST,
                               const float* __restrict__ g, const float* __restrict__ beta,
                               int co, int cbase, float* __restrict__ FEATB) {
  int gid = blockIdx.x * 256 + threadIdx.x;          // B*co*N
  int n = gid & (NN - 1);
  int t = gid >> 12;
  int o = t % co, b = t / co;
  const float cnt = (float)BB * NN;
  float m = BNST[o * 2] / cnt;
  float var = BNST[o * 2 + 1] / cnt - m * m;
  float v = outpre[((size_t)(b * co + o)) * NN + n];
  float y = (v - m) * rsqrtf(var + EPSV) * g[o] + beta[o];
  FEATB[((size_t)(b * 224 + cbase + o)) * NN + n] = fmaxf(y, 0.f);
}

// =================== FPS (argmax dist from point 0) + negd2 =================
__global__ void fps_kernel(const float* __restrict__ X, int* __restrict__ SIDX) {
  int b = blockIdx.x, tid = threadIdx.x;
  const float* xb = X + (size_t)b * 3 * NN;
  float x00 = xb[0], x01 = xb[NN], x02 = xb[2 * NN];
  float best = -1.f; int bi = 0;
  for (int n = tid; n < NN; n += 256) {
    float d0 = xb[n] - x00, d1 = xb[NN + n] - x01, d2 = xb[2 * NN + n] - x02;
    float dd = d0 * d0 + d1 * d1 + d2 * d2;
    if (dd > best) { best = dd; bi = n; }
  }
  __shared__ float rv[256]; __shared__ int ri[256];
  rv[tid] = best; ri[tid] = bi; __syncthreads();
  for (int s = 128; s > 0; s >>= 1) {
    if (tid < s) {
      if (rv[tid + s] > rv[tid] || (rv[tid + s] == rv[tid] && ri[tid + s] < ri[tid])) {
        rv[tid] = rv[tid + s]; ri[tid] = ri[tid + s];
      }
    }
    __syncthreads();
  }
  if (tid == 0) SIDX[b] = ri[0];
}

__global__ void negd2_kernel(const float* __restrict__ X, const int* __restrict__ SIDX,
                             float* __restrict__ NEGD2) {
  int gid = blockIdx.x * 256 + threadIdx.x;          // B*2*N
  int n = gid & (NN - 1);
  int t = gid >> 12;
  int s = t & 1, b = t >> 1;
  const float* xb = X + (size_t)b * 3 * NN;
  int p = s ? SIDX[b] : 0;
  float d0 = xb[n] - xb[p], d1 = xb[NN + n] - xb[NN + p], d2 = xb[2 * NN + n] - xb[2 * NN + p];
  NEGD2[gid] = -(d0 * d0 + d1 * d1 + d2 * d2);
}

// ======= exact rank-(N/2) threshold via value bisection (deterministic) =====
__global__ void thresh_kernel(const float* __restrict__ NEGD2, float* __restrict__ TH) {
  int bs = blockIdx.x, tid = threadIdx.x;
  const float* rowg = NEGD2 + (size_t)bs * NN;
  __shared__ float rv[NN];
  __shared__ float red[256];
  for (int i = tid; i < NN; i += 256) rv[i] = rowg[i];
  __syncthreads();

  float mn = 1e30f;
  for (int i = tid; i < NN; i += 256) mn = fminf(mn, rv[i]);
  red[tid] = mn; __syncthreads();
  for (int s = 128; s > 0; s >>= 1) { if (tid < s) red[tid] = fminf(red[tid], red[tid + s]); __syncthreads(); }
  float lo = red[0] - 1.f, hi = 1e-6f;
  __syncthreads();

  for (int it = 0; it < 48; ++it) {                  // interval shrinks below float ULP
    float mid = 0.5f * (lo + hi);
    int cnt = 0;
    for (int i = tid; i < NN; i += 256) cnt += (rv[i] >= mid) ? 1 : 0;
    red[tid] = (float)cnt; __syncthreads();
    for (int s = 128; s > 0; s >>= 1) { if (tid < s) red[tid] += red[tid + s]; __syncthreads(); }
    int total = (int)red[0];
    __syncthreads();
    if (total >= NN / 2) lo = mid; else hi = mid;
  }
  // snap: t = largest data value < hi  (== the rank-N/2 value)
  float mx = -1e30f;
  for (int i = tid; i < NN; i += 256) if (rv[i] < hi) mx = fmaxf(mx, rv[i]);
  red[tid] = mx; __syncthreads();
  for (int s = 128; s > 0; s >>= 1) { if (tid < s) red[tid] = fmaxf(red[tid], red[tid + s]); __syncthreads(); }
  float t = red[0];
  __syncthreads();

  int ngt = 0, neq = 0;
  for (int i = tid; i < NN; i += 256) { ngt += (rv[i] > t) ? 1 : 0; neq += (rv[i] == t) ? 1 : 0; }
  red[tid] = (float)ngt; __syncthreads();
  for (int s = 128; s > 0; s >>= 1) { if (tid < s) red[tid] += red[tid + s]; __syncthreads(); }
  float fngt = red[0]; __syncthreads();
  red[tid] = (float)neq; __syncthreads();
  for (int s = 128; s > 0; s >>= 1) { if (tid < s) red[tid] += red[tid + s]; __syncthreads(); }
  float fneq = red[0];
  if (tid == 0) { TH[bs * 4] = t; TH[bs * 4 + 1] = fngt; TH[bs * 4 + 2] = fneq; }
}

// ================= masked mean over selected N/2 points -> h0 ===============
__global__ void pool_kernel(const float* __restrict__ NEGD2, const float* __restrict__ TH,
                            const float* __restrict__ FEATB, float* __restrict__ H0) {
  int bs = blockIdx.x, tid = threadIdx.x;
  int b = bs >> 1, s = bs & 1;
  float t = TH[bs * 4], ngt = TH[bs * 4 + 1], neq = TH[bs * 4 + 2];
  float scale = ((float)(NN / 2) - ngt) / neq;       // fraction of tied elems taken
  __shared__ float rv[NN];
  __shared__ float red[256];
  const float* rowg = NEGD2 + (size_t)bs * NN;
  for (int i = tid; i < NN; i += 256) rv[i] = rowg[i];
  __syncthreads();
  for (int c = 0; c < 224; ++c) {
    const float* f = FEATB + ((size_t)(b * 224 + c)) * NN;
    float sgt = 0.f, seq = 0.f;
    for (int i = tid; i < NN; i += 256) {
      float v = rv[i], fv = f[i];
      if (v > t) sgt += fv; else if (v == t) seq += fv;
    }
    red[tid] = sgt; __syncthreads();
    for (int st = 128; st > 0; st >>= 1) { if (tid < st) red[tid] += red[tid + st]; __syncthreads(); }
    float Sgt = red[0]; __syncthreads();
    red[tid] = seq; __syncthreads();
    for (int st = 128; st > 0; st >>= 1) { if (tid < st) red[tid] += red[tid + st]; __syncthreads(); }
    float Seq = red[0]; __syncthreads();
    if (tid == 0) H0[b * 448 + c * 2 + s] = (Sgt + Seq * scale) * (2.0f / NN);
  }
}

// ===========================================================================
extern "C" void kernel_launch(void* const* d_in, const int* in_sizes, int n_in,
                              void* d_out, int out_size, void* d_ws, size_t ws_size,
                              hipStream_t stream) {
  (void)in_sizes; (void)n_in; (void)out_size; (void)ws_size;
  auto IN = [&](int i) { return (const float*)d_in[i]; };
  const float* X = IN(0);

  // Param flattening (jax tree order = sorted dict keys). Per c-layer (14):
  //  0:bn_beta 1:bn_g 2:dn_b1 3:dn_b2 4:dn_beta 5:dn_g 6:dn_w1 7:dn_w2
  //  8:lin_b 9:lin_w 10:wn_b 11:wn_beta 12:wn_g 13:wn_w
  // Tail: 43:l1_b 44:l1_w 45:l2_b 46:l2_w 47:l3_b 48:l3_w
  const int PERL = 14;
  const int ci_a[3]      = {3, 32, 64};
  const int co_a[3]      = {32, 64, 128};
  const int cbase_in[3]  = {0, 0, 32};
  const int cbase_out[3] = {0, 32, 96};

  // ---------------- workspace carve-out ----------------
  char* wsb = (char*)d_ws;
  size_t off = 0;
  auto take = [&](size_t bytes) -> char* {
    char* p = wsb + off;
    off = (off + bytes + 255) & ~(size_t)255;
    return p;
  };
  float*     MOMP  = (float*)take(64 * 16 * 4);
  float*     MOM   = (float*)take(16 * 4);
  float*     LC    = (float*)take(3 * 256 * 4);
  float*     SA    = (float*)take((size_t)BB * NN * 4);
  int*       NNI   = (int*)  take((size_t)BB * NN * KNN * 4);
  float*     INVD  = (float*)take((size_t)BB * NN * 4);
  float*     SRW   = (float*)take((size_t)BB * KNN * NN * 4);
  float*     LNN   = (float*)take((size_t)BB * 3 * KNN * NN * 4);
  float*     FEATB = (float*)take((size_t)BB * 224 * NN * 4);
  float*     OUTP  = (float*)take((size_t)BB * 128 * NN * 4);
  float*     BNST  = (float*)take(256 * 4);
  _Float16*  F16   = (_Float16*)take((size_t)16384 * 2048 * 2);
  _Float16*  W16   = (_Float16*)take((size_t)262144 * 2);
  _Float16*  A16   = (_Float16*)take((size_t)16 * 512 * 2);
  float*     H0    = (float*)take(4 * 448 * 4);
  float*     H1    = (float*)take(16 * 512 * 4);
  float*     H2    = (float*)take(16 * 256 * 4);
  float*     ND2   = (float*)take((size_t)BB * 2 * NN * 4);
  float*     TH    = (float*)take(BB * 2 * 4 * 4);
  int*       SIDX  = (int*)  take(BB * 4);

  // ---------------- kNN + density + geometry ----------------
  sa_kernel    <<<BB * NN / 256, 256, 0, stream>>>(X, SA);
  knn_kernel   <<<BB * NN / 8,   256, 0, stream>>>(X, SA, NNI, INVD);
  prep_kernel  <<<BB * NN / 256, 256, 0, stream>>>(X, NNI, INVD, SRW, LNN, MOMP);
  momsum_kernel<<<1, 16, 0, stream>>>(MOMP, MOM);

  K3P p3;
  for (int li = 0; li < 3; ++li) {
    int base = 1 + PERL * li;
    p3.dn_w1[li]   = IN(base + 6);
    p3.dn_g[li]    = IN(base + 5);
    p3.dn_beta[li] = IN(base + 4);
    p3.wn_w[li]    = IN(base + 13);
    p3.wn_g[li]    = IN(base + 12);
    p3.wn_beta[li] = IN(base + 11);
  }
  lconst_kernel<<<1, 32, 0, stream>>>(MOM, LC, p3);

  // ---------------- three point-conv layers ----------------
  for (int li = 0; li < 3; ++li) {
    int base = 1 + PERL * li;
    const float* dn_w2   = IN(base + 7);
    const float* dn_b2   = IN(base + 3);
    const float* lin_b   = IN(base + 8);
    const float* lin_w   = IN(base + 9);
    const float* bn_g    = IN(base + 1);
    const float* bn_beta = IN(base + 0);
    int ci = ci_a[li], co = co_a[li], ciw = ci * 32;
    const float* xin = (li == 0) ? X : (FEATB + (size_t)cbase_in[li] * NN);
    int bstride = (li == 0) ? 3 * NN : 224 * NN;

    feat_kernel<<<BB * NN / 8, 256, 0, stream>>>(xin, bstride, ci, NNI, SRW, LNN,
                                                 LC + li * 256, dn_w2, dn_b2, F16);
    w16_kernel<<<(co * ciw + 255) / 256, 256, 0, stream>>>(lin_w, W16, co, co, ciw);
    int tiles = (16384 / 16) * (co / 16);
    gemm16_kernel<<<tiles / 8, 256, 0, stream>>>(F16, W16, lin_b, OUTP,
                                                 ciw, co, co, 16384, 16384, 1, 0, NN, co);
    bnstat_kernel<<<co, 256, 0, stream>>>(OUTP, co, BNST);
    bnapply_kernel<<<BB * co * NN / 256, 256, 0, stream>>>(OUTP, BNST, bn_g, bn_beta,
                                                           co, cbase_out[li], FEATB);
  }

  // ---------------- pooling path ----------------
  fps_kernel   <<<BB, 256, 0, stream>>>(X, SIDX);
  negd2_kernel <<<BB * 2 * NN / 256, 256, 0, stream>>>(X, SIDX, ND2);
  thresh_kernel<<<BB * 2, 256, 0, stream>>>(ND2, TH);
  pool_kernel  <<<BB * 2, 256, 0, stream>>>(ND2, TH, FEATB, H0);

  // ---------------- final MLP (WMMA, M padded 4->16) ----------------
  const float* l1_b = IN(1 + PERL * 3 + 0);
  const float* l1_w = IN(1 + PERL * 3 + 1);
  const float* l2_b = IN(1 + PERL * 3 + 2);
  const float* l2_w = IN(1 + PERL * 3 + 3);
  const float* l3_b = IN(1 + PERL * 3 + 4);
  const float* l3_w = IN(1 + PERL * 3 + 5);

  act16_kernel<<<(16 * 448) / 256, 256, 0, stream>>>(H0, A16, 4, 448, 448);
  w16_kernel  <<<(512 * 448 + 255) / 256, 256, 0, stream>>>(l1_w, W16, 512, 512, 448);
  gemm16_kernel<<<4, 256, 0, stream>>>(A16, W16, l1_b, H1, 448, 512, 512, 16, 4, 0, 512, 1, 0);

  act16_kernel<<<(16 * 512) / 256, 256, 0, stream>>>(H1, A16, 4, 512, 512);
  w16_kernel  <<<(256 * 512 + 255) / 256, 256, 0, stream>>>(l2_w, W16, 256, 256, 512);
  gemm16_kernel<<<2, 256, 0, stream>>>(A16, W16, l2_b, H2, 512, 256, 256, 16, 4, 0, 256, 1, 0);

  act16_kernel<<<(16 * 256) / 256, 256, 0, stream>>>(H2, A16, 4, 256, 256);
  w16_kernel  <<<(48 * 256 + 255) / 256, 256, 0, stream>>>(l3_w, W16, 40, 48, 256);
  gemm16_kernel<<<1, 256, 0, stream>>>(A16, W16, l3_b, (float*)d_out, 256, 40, 48, 16, 4, 0, 40, 1, 0);
}
